// PointNetSetAbstractionMsg_84851373899984
// MI455X (gfx1250) — matrix operand order repaired and checked by reading, not compiled, not run on hardware
//
#include <hip/hip_runtime.h>

typedef __attribute__((ext_vector_type(16))) _Float16 v16h;
typedef __attribute__((ext_vector_type(8)))  _Float16 v8h;
typedef __attribute__((ext_vector_type(8)))  float    v8f;

#define NPTS 8192
#define NB   8
#define NS   1024
#define STR  104   // LDS row stride in halves: 208B = 13*16B (b128-aligned, bank-friendly)

// ---------------------------------------------------------------------------
// build full = concat(xyz_t, points_t): (B, N, 8)
// ---------------------------------------------------------------------------
__global__ __launch_bounds__(256) void build_full(const float* __restrict__ xyz,
                                                  const float* __restrict__ pts,
                                                  float* __restrict__ full) {
  int i = blockIdx.x * 256 + threadIdx.x;
  if (i >= NB * NPTS) return;
  int b = i >> 13, n = i & (NPTS - 1);
  float* o = full + (size_t)i * 8;
  o[0] = xyz[(size_t)b * 2 * NPTS + n];
  o[1] = xyz[(size_t)b * 2 * NPTS + NPTS + n];
#pragma unroll
  for (int c = 0; c < 6; c++) o[2 + c] = pts[((size_t)b * 6 + c) * NPTS + n];
}

// ---------------------------------------------------------------------------
// Farthest point sampling: 1 block per batch, 256 threads, 32 pts/thread in regs
// Emits new_xyz (B,S,2) to ws and the transposed new_xyz output directly.
// ---------------------------------------------------------------------------
__global__ __launch_bounds__(256) void fps_kernel(const float* __restrict__ xyz,
                                                  float* __restrict__ nxyz,
                                                  float* __restrict__ out) {
  int b = blockIdx.x, tid = threadIdx.x, lane = tid & 31, w = tid >> 5;
  const float* X = xyz + (size_t)b * 2 * NPTS;
  const float* Y = X + NPTS;
  float px[32], py[32], dist[32];
#pragma unroll
  for (int j = 0; j < 32; j++) {
    int n = tid + j * 256;
    px[j] = X[n]; py[j] = Y[n]; dist[j] = 1e10f;
  }
  __shared__ float cent[2];
  __shared__ int   sfar;
  __shared__ float wval[8];
  __shared__ int   widx[8];
  if (tid == 0) sfar = 0;
  __syncthreads();
  for (int t = 0; t < NS; t++) {
    if (tid == 0) {
      int f = sfar;
      float cx = X[f], cy = Y[f];
      cent[0] = cx; cent[1] = cy;
      nxyz[((size_t)b * NS + t) * 2 + 0] = cx;
      nxyz[((size_t)b * NS + t) * 2 + 1] = cy;
      out[(size_t)b * 2 * NS + t]      = cx;   // (B,2,S) channel 0
      out[(size_t)b * 2 * NS + NS + t] = cy;   // channel 1
    }
    __syncthreads();
    float cx = cent[0], cy = cent[1];
    float best = -1.0f; int bi = 0;
#pragma unroll
    for (int j = 0; j < 32; j++) {
      float dx = px[j] - cx, dy = py[j] - cy;
      float d = __fmul_rn(dx, dx) + __fmul_rn(dy, dy);
      float nd = dist[j] < d ? dist[j] : d;
      dist[j] = nd;
      int n = tid + j * 256;
      if (nd > best || (nd == best && n < bi)) { best = nd; bi = n; }
    }
#pragma unroll
    for (int off = 16; off > 0; off >>= 1) {
      float ov = __shfl_down(best, off);
      int   oi = __shfl_down(bi, off);
      if (ov > best || (ov == best && oi < bi)) { best = ov; bi = oi; }
    }
    if (lane == 0) { wval[w] = best; widx[w] = bi; }
    __syncthreads();
    if (tid == 0) {
      float bb = wval[0]; int ii = widx[0];
#pragma unroll
      for (int k = 1; k < 8; k++)
        if (wval[k] > bb || (wval[k] == bb && widx[k] < ii)) { bb = wval[k]; ii = widx[k]; }
      sfar = ii;
    }
    __syncthreads();
  }
}

// ---------------------------------------------------------------------------
// Ball query + gather + recenter: 1 wave per (b,s); ordered append of first K
// in-radius indices; pad with first member. grouped: (B*S, K, 8) f32
// ---------------------------------------------------------------------------
template <int K>
__global__ __launch_bounds__(256) void qball_group(const float* __restrict__ full,
                                                   const float* __restrict__ nxyz,
                                                   float* __restrict__ grouped, float r2) {
  int wid = blockIdx.x * 8 + (threadIdx.x >> 5);
  int lane = threadIdx.x & 31;
  int b = wid >> 10;
  const float* q = nxyz + (size_t)wid * 2;
  float qx = q[0], qy = q[1];
  float qs = __fmul_rn(qx, qx) + __fmul_rn(qy, qy);
  const float* F = full + (size_t)b * NPTS * 8;
  float* G = grouped + (size_t)wid * K * 8;
  int count = 0, firstI = -1;
  for (int base = 0; base < NPTS && count < K; base += 32) {
    int n = base + lane;
    float pxv = F[(size_t)n * 8 + 0], pyv = F[(size_t)n * 8 + 1];
    float ps  = __fmul_rn(pxv, pxv) + __fmul_rn(pyv, pyv);
    float dot = __fmul_rn(qx, pxv) + __fmul_rn(qy, pyv);
    float sq  = (qs + ps) - 2.0f * dot;
    bool in = (sq <= r2);
    unsigned mask = (unsigned)__ballot(in);
    if (firstI < 0 && mask) firstI = base + (__ffs(mask) - 1);
    int pre = __popc(mask & ((1u << lane) - 1u));
    int pos = count + pre;
    if (in && pos < K) {
      float* o = G + (size_t)pos * 8;
      o[0] = pxv - qx; o[1] = pyv - qy;
#pragma unroll
      for (int c = 2; c < 8; c++) o[c] = F[(size_t)n * 8 + c];
    }
    count += __popc(mask);
  }
  int filled = count < K ? count : K;
  if (firstI < 0) firstI = 0;
  float f0 = F[(size_t)firstI * 8 + 0] - qx, f1 = F[(size_t)firstI * 8 + 1] - qy;
  for (int p = filled + lane; p < K; p += 32) {
    float* o = G + (size_t)p * 8;
    o[0] = f0; o[1] = f1;
#pragma unroll
    for (int c = 2; c < 8; c++) o[c] = F[(size_t)firstI * 8 + c];
  }
}

// ---------------------------------------------------------------------------
// WMMA fragment loaders (ISA 7.12.2 layouts, 16x16x32 f16)
// A (16x32 MxK): lane<16 -> K {kk..kk+7, kk+16..kk+23}; lane>=16 -> +8
// B (32x16 KxN) staged as wT[n][k]: lane holds col n=lane&15, same K split as A
// ---------------------------------------------------------------------------
__device__ inline v16h load_A(const _Float16* base, int mb, int kk, int lane) {
  const _Float16* p = base + (size_t)(mb + (lane & 15)) * STR + kk + ((lane & 16) ? 8 : 0);
  v8h lo = *reinterpret_cast<const v8h*>(p);
  v8h hi = *reinterpret_cast<const v8h*>(p + 16);
  v16h r;
#pragma unroll
  for (int i = 0; i < 8; i++) { r[i] = lo[i]; r[i + 8] = hi[i]; }
  return r;
}
__device__ inline v16h load_B(const _Float16* base, int nb, int kk, int lane) {
  const _Float16* p = base + (size_t)(nb + (lane & 15)) * STR + kk + ((lane & 16) ? 8 : 0);
  v8h lo = *reinterpret_cast<const v8h*>(p);
  v8h hi = *reinterpret_cast<const v8h*>(p + 16);
  v16h r;
#pragma unroll
  for (int i = 0; i < 8; i++) { r[i] = lo[i]; r[i + 8] = hi[i]; }
  return r;
}

// MODE: 0 = affine+relu -> store f16 to outb; 1 = raw-y stats; 2 = affine+relu+colmax
template <int K, int CINP, int COUT, int MODE, int CMO>
__device__ inline void layer_compute(const _Float16* in, _Float16* outb, const _Float16* wl,
                                     const float* bias, const float* aff,
                                     float* wsum, float* wsq, unsigned* cmax,
                                     int lane, int w) {
  constexpr int MT = K / 16, NT = COUT / 16, TILES = MT * NT;
  for (int t = w; t < TILES; t += 8) {
    int mt = t / NT, nt = t - mt * NT;
    int mb = mt * 16, nb = nt * 16;
    int n = nb + (lane & 15);
    float bv = bias[n];
    v8f acc;
#pragma unroll
    for (int r = 0; r < 8; r++) acc[r] = bv;
#pragma unroll
    for (int kk = 0; kk < CINP; kk += 32) {
      v16h a  = load_A(in, mb, kk, lane);
      v16h bm = load_B(wl, nb, kk, lane);
      acc = __builtin_amdgcn_wmma_f32_16x16x32_f16(false, a, false, bm, (short)0, acc,
                                                   false, false);
    }
    if constexpr (MODE == 1) {
      float s1 = 0.f, s2 = 0.f;
#pragma unroll
      for (int r = 0; r < 8; r++) { s1 += acc[r]; s2 += acc[r] * acc[r]; }
      s1 += __shfl_xor(s1, 16);
      s2 += __shfl_xor(s2, 16);
      if (lane < 16) { wsum[w * CMO + n] += s1; wsq[w * CMO + n] += s2; }
    } else if constexpr (MODE == 0) {
      float sc = aff[2 * n], sh = aff[2 * n + 1];
#pragma unroll
      for (int r = 0; r < 8; r++) {
        float z = acc[r] * sc + sh; z = z > 0.f ? z : 0.f;
        int m = mb + r + ((lane & 16) ? 8 : 0);
        outb[(size_t)m * STR + n] = (_Float16)z;
      }
    } else {
      float sc = aff[2 * n], sh = aff[2 * n + 1];
      float mv = 0.f;
#pragma unroll
      for (int r = 0; r < 8; r++) {
        float z = acc[r] * sc + sh; z = z > 0.f ? z : 0.f;
        mv = z > mv ? z : mv;
      }
      float o = __shfl_xor(mv, 16);
      mv = o > mv ? o : mv;
      if (lane < 16) atomicMax(cmax + n, __float_as_uint(mv));
    }
  }
}

// ---------------------------------------------------------------------------
// Fused MLP chain, one WG per (b,s). PASS=1..3: compute layers 1..PASS (known
// BN affines for earlier layers), emit per-WG sum/sumsq partials for layer PASS.
// PASS=4: full chain + max over K -> output.
// ---------------------------------------------------------------------------
template <int K, int C1, int C2, int C3, int PASS, int CH_OFF>
__global__ __launch_bounds__(256) void mlp_pass(
    const float* __restrict__ grouped,
    const float* __restrict__ W1, const float* __restrict__ B1,
    const float* __restrict__ W2, const float* __restrict__ B2,
    const float* __restrict__ W3, const float* __restrict__ B3,
    const float* __restrict__ A1, const float* __restrict__ A2, const float* __restrict__ A3,
    float* __restrict__ partials, float* __restrict__ outp) {
  constexpr int CMO = (C1 > C2 ? (C1 > C3 ? C1 : C3) : (C2 > C3 ? C2 : C3));
  __shared__ _Float16 actA[K * STR];
  __shared__ _Float16 actB[K * STR];
  __shared__ _Float16 wlds[CMO * STR];
  __shared__ float    wsum[8 * CMO];
  __shared__ float    wsq[8 * CMO];
  __shared__ unsigned cmax[CMO];
  int bs = blockIdx.x, tid = threadIdx.x, lane = tid & 31, w = tid >> 5;

  if constexpr (PASS < 4) {
    for (int i = tid; i < 8 * CMO; i += 256) { wsum[i] = 0.f; wsq[i] = 0.f; }
  } else {
    for (int i = tid; i < CMO; i += 256) cmax[i] = 0u;
  }
  // grouped (K x 8) -> actA f16, zero-padded to 32 input channels
  const float* g = grouped + (size_t)bs * (K * 8);
  for (int i = tid; i < K * 32; i += 256) {
    int m = i >> 5, c = i & 31;
    actA[(size_t)m * STR + c] = (_Float16)((c < 8) ? g[(size_t)m * 8 + c] : 0.f);
  }
  // W1 (C1 x 8) -> wlds, zero-padded to 32
  for (int i = tid; i < C1 * 32; i += 256) {
    int o = i >> 5, c = i & 31;
    wlds[(size_t)o * STR + c] = (_Float16)((c < 8) ? W1[(size_t)o * 8 + c] : 0.f);
  }
  __syncthreads();
  // layer 1
  if constexpr (PASS == 1)
    layer_compute<K, 32, C1, 1, CMO>(actA, nullptr, wlds, B1, nullptr, wsum, wsq, nullptr, lane, w);
  else
    layer_compute<K, 32, C1, 0, CMO>(actA, actB, wlds, B1, A1, nullptr, nullptr, nullptr, lane, w);
  __syncthreads();
  // layer 2
  if constexpr (PASS >= 2) {
    for (int i = tid; i < C2 * C1; i += 256) {
      int o = i / C1, c = i - o * C1;
      wlds[(size_t)o * STR + c] = (_Float16)W2[(size_t)o * C1 + c];
    }
    __syncthreads();
    if constexpr (PASS == 2)
      layer_compute<K, C1, C2, 1, CMO>(actB, nullptr, wlds, B2, nullptr, wsum, wsq, nullptr, lane, w);
    else
      layer_compute<K, C1, C2, 0, CMO>(actB, actA, wlds, B2, A2, nullptr, nullptr, nullptr, lane, w);
    __syncthreads();
  }
  // layer 3
  if constexpr (PASS >= 3) {
    for (int i = tid; i < C3 * C2; i += 256) {
      int o = i / C2, c = i - o * C2;
      wlds[(size_t)o * STR + c] = (_Float16)W3[(size_t)o * C2 + c];
    }
    __syncthreads();
    if constexpr (PASS == 3)
      layer_compute<K, C2, C3, 1, CMO>(actA, nullptr, wlds, B3, nullptr, wsum, wsq, nullptr, lane, w);
    else
      layer_compute<K, C2, C3, 2, CMO>(actA, nullptr, wlds, B3, A3, nullptr, nullptr, cmax, lane, w);
    __syncthreads();
  }
  // epilogue
  if constexpr (PASS < 4) {
    constexpr int CS = (PASS == 1) ? C1 : ((PASS == 2) ? C2 : C3);
    if (tid < CS) {
      float s1 = 0.f, s2 = 0.f;
#pragma unroll
      for (int ww = 0; ww < 8; ww++) { s1 += wsum[ww * CMO + tid]; s2 += wsq[ww * CMO + tid]; }
      partials[(size_t)bs * (2 * CS) + tid]      = s1;
      partials[(size_t)bs * (2 * CS) + CS + tid] = s2;
    }
  } else {
    if (tid < C3) {
      int b = bs >> 10, s = bs & (NS - 1);
      outp[((size_t)(b * 320 + CH_OFF + tid)) * NS + s] = __uint_as_float(cmax[tid]);
    }
  }
}

// ---------------------------------------------------------------------------
// Deterministic global reduction of per-WG partials -> BN affine (scale, shift)
// ---------------------------------------------------------------------------
__global__ __launch_bounds__(256) void finalize_stats(const float* __restrict__ partials,
                                                      const float* __restrict__ gamma,
                                                      const float* __restrict__ beta,
                                                      float* __restrict__ aff, int C,
                                                      float invN) {
  int c = blockIdx.x, tid = threadIdx.x;
  float s1 = 0.f, s2 = 0.f;
  for (int wg = tid; wg < NB * NS; wg += 256) {
    s1 += partials[(size_t)wg * 2 * C + c];
    s2 += partials[(size_t)wg * 2 * C + C + c];
  }
  __shared__ float r1[256], r2[256];
  r1[tid] = s1; r2[tid] = s2;
  __syncthreads();
  for (int off = 128; off > 0; off >>= 1) {
    if (tid < off) { r1[tid] += r1[tid + off]; r2[tid] += r2[tid + off]; }
    __syncthreads();
  }
  if (tid == 0) {
    float mean = r1[0] * invN;
    float var  = r2[0] * invN - mean * mean;
    float sc   = gamma[c] / sqrtf(var + 1e-5f);
    aff[2 * c]     = sc;
    aff[2 * c + 1] = beta[c] - mean * sc;
  }
}

// ---------------------------------------------------------------------------
// Host launcher
// ---------------------------------------------------------------------------
extern "C" void kernel_launch(void* const* d_in, const int* in_sizes, int n_in,
                              void* d_out, int out_size, void* d_ws, size_t ws_size,
                              hipStream_t stream) {
  (void)in_sizes; (void)n_in; (void)out_size; (void)ws_size;
  const float* xyz = (const float*)d_in[0];
  const float* pts = (const float*)d_in[1];
  // params pytree order: per (scale, layer) dict keys sorted: b, beta, gamma, w
  auto prm = [&](int sc, int ly, int which) -> const float* {
    return (const float*)d_in[2 + (sc * 3 + ly) * 4 + which];
  };
  float* out  = (float*)d_out;
  float* outp = out + NB * 2 * NS;  // second tuple element, (B,320,S)

  float* full     = (float*)d_ws;                // 8*8192*8       = 524288 f
  float* nxyz     = full + 524288;               // 8*1024*2       = 16384 f
  float* grouped  = nxyz + 16384;                // 8*1024*128*8   = 8388608 f
  float* partials = grouped + 8388608;           // 8192*256       = 2097152 f
  float* aff      = partials + 2097152;          // 3*256 floats
  float* aff2     = aff + 256;
  float* aff3     = aff + 512;

  build_full<<<(NB * NPTS + 255) / 256, 256, 0, stream>>>(xyz, pts, full);
  fps_kernel<<<NB, 256, 0, stream>>>(xyz, nxyz, out);

  const int NWG = NB * NS;  // 8192 (b,s) workgroups

  // ---- scale 0: K=16, MLP 8->32->32->64, r=0.1 ----
  {
    float r2 = (float)(0.1 * 0.1);
    float invN = (float)(1.0 / (8.0 * 1024.0 * 16.0));
    qball_group<16><<<NWG / 8, 256, 0, stream>>>(full, nxyz, grouped, r2);
    mlp_pass<16, 32, 32, 64, 1, 0><<<NWG, 256, 0, stream>>>(grouped,
        prm(0,0,3), prm(0,0,0), prm(0,1,3), prm(0,1,0), prm(0,2,3), prm(0,2,0),
        aff, aff2, aff3, partials, outp);
    finalize_stats<<<32, 256, 0, stream>>>(partials, prm(0,0,2), prm(0,0,1), aff,  32, invN);
    mlp_pass<16, 32, 32, 64, 2, 0><<<NWG, 256, 0, stream>>>(grouped,
        prm(0,0,3), prm(0,0,0), prm(0,1,3), prm(0,1,0), prm(0,2,3), prm(0,2,0),
        aff, aff2, aff3, partials, outp);
    finalize_stats<<<32, 256, 0, stream>>>(partials, prm(0,1,2), prm(0,1,1), aff2, 32, invN);
    mlp_pass<16, 32, 32, 64, 3, 0><<<NWG, 256, 0, stream>>>(grouped,
        prm(0,0,3), prm(0,0,0), prm(0,1,3), prm(0,1,0), prm(0,2,3), prm(0,2,0),
        aff, aff2, aff3, partials, outp);
    finalize_stats<<<64, 256, 0, stream>>>(partials, prm(0,2,2), prm(0,2,1), aff3, 64, invN);
    mlp_pass<16, 32, 32, 64, 4, 0><<<NWG, 256, 0, stream>>>(grouped,
        prm(0,0,3), prm(0,0,0), prm(0,1,3), prm(0,1,0), prm(0,2,3), prm(0,2,0),
        aff, aff2, aff3, partials, outp);
  }

  // ---- scale 1: K=32, MLP 8->64->64->128, r=0.2 ----
  {
    float r2 = (float)(0.2 * 0.2);
    float invN = (float)(1.0 / (8.0 * 1024.0 * 32.0));
    qball_group<32><<<NWG / 8, 256, 0, stream>>>(full, nxyz, grouped, r2);
    mlp_pass<32, 64, 64, 128, 1, 64><<<NWG, 256, 0, stream>>>(grouped,
        prm(1,0,3), prm(1,0,0), prm(1,1,3), prm(1,1,0), prm(1,2,3), prm(1,2,0),
        aff, aff2, aff3, partials, outp);
    finalize_stats<<<64, 256, 0, stream>>>(partials, prm(1,0,2), prm(1,0,1), aff,  64, invN);
    mlp_pass<32, 64, 64, 128, 2, 64><<<NWG, 256, 0, stream>>>(grouped,
        prm(1,0,3), prm(1,0,0), prm(1,1,3), prm(1,1,0), prm(1,2,3), prm(1,2,0),
        aff, aff2, aff3, partials, outp);
    finalize_stats<<<64, 256, 0, stream>>>(partials, prm(1,1,2), prm(1,1,1), aff2, 64, invN);
    mlp_pass<32, 64, 64, 128, 3, 64><<<NWG, 256, 0, stream>>>(grouped,
        prm(1,0,3), prm(1,0,0), prm(1,1,3), prm(1,1,0), prm(1,2,3), prm(1,2,0),
        aff, aff2, aff3, partials, outp);
    finalize_stats<<<128, 256, 0, stream>>>(partials, prm(1,2,2), prm(1,2,1), aff3, 128, invN);
    mlp_pass<32, 64, 64, 128, 4, 64><<<NWG, 256, 0, stream>>>(grouped,
        prm(1,0,3), prm(1,0,0), prm(1,1,3), prm(1,1,0), prm(1,2,3), prm(1,2,0),
        aff, aff2, aff3, partials, outp);
  }

  // ---- scale 2: K=128, MLP 8->64->96->128, r=0.4 ----
  {
    float r2 = (float)(0.4 * 0.4);
    float invN = (float)(1.0 / (8.0 * 1024.0 * 128.0));
    qball_group<128><<<NWG / 8, 256, 0, stream>>>(full, nxyz, grouped, r2);
    mlp_pass<128, 64, 96, 128, 1, 192><<<NWG, 256, 0, stream>>>(grouped,
        prm(2,0,3), prm(2,0,0), prm(2,1,3), prm(2,1,0), prm(2,2,3), prm(2,2,0),
        aff, aff2, aff3, partials, outp);
    finalize_stats<<<64, 256, 0, stream>>>(partials, prm(2,0,2), prm(2,0,1), aff,  64, invN);
    mlp_pass<128, 64, 96, 128, 2, 192><<<NWG, 256, 0, stream>>>(grouped,
        prm(2,0,3), prm(2,0,0), prm(2,1,3), prm(2,1,0), prm(2,2,3), prm(2,2,0),
        aff, aff2, aff3, partials, outp);
    finalize_stats<<<96, 256, 0, stream>>>(partials, prm(2,1,2), prm(2,1,1), aff2, 96, invN);
    mlp_pass<128, 64, 96, 128, 3, 192><<<NWG, 256, 0, stream>>>(grouped,
        prm(2,0,3), prm(2,0,0), prm(2,1,3), prm(2,1,0), prm(2,2,3), prm(2,2,0),
        aff, aff2, aff3, partials, outp);
    finalize_stats<<<128, 256, 0, stream>>>(partials, prm(2,2,2), prm(2,2,1), aff3, 128, invN);
    mlp_pass<128, 64, 96, 128, 4, 192><<<NWG, 256, 0, stream>>>(grouped,
        prm(2,0,3), prm(2,0,0), prm(2,1,3), prm(2,1,0), prm(2,2,3), prm(2,2,0),
        aff, aff2, aff3, partials, outp);
  }
}
